// ComplexModel_64003602645116
// MI455X (gfx1250) — matrix operand order repaired and checked by reading, not compiled, not run on hardware
//
#include <hip/hip_runtime.h>
#include <hip/hip_bf16.h>

#define Bn 64
#define Tn 256
#define Dn 512
#define Hn 1024
#define Gn (4 * Hn)

typedef __bf16 bf16_t;
typedef __attribute__((ext_vector_type(16))) __bf16 v16bf;
typedef __attribute__((ext_vector_type(8)))  __bf16 v8bf;
typedef __attribute__((ext_vector_type(8)))  float  v8f;

// ---------------------------------------------------------------------------
// WMMA fragment loaders (layouts per CDNA5 ISA 7.12.2, 16-bit A 16x32 / B 32x16)
// A: lane m=lane%16, half=lane/16; elems 0..7 = K[8*half .. 8*half+7],
//    elems 8..15 = K[16+8*half .. 16+8*half+7]
// B (stored transposed, N x K row-major): lane n=lane%16; elems 0..7 =
//    K[8*half .. 8*half+7], elems 8..15 = K[16+8*half .. 16+8*half+7]
// ---------------------------------------------------------------------------
__device__ __forceinline__ v16bf load_a_frag(const bf16_t* __restrict__ base,
                                             int stride, int row, int lane) {
  int r = row + (lane & 15);
  int half = lane >> 4;
  const bf16_t* p = base + (size_t)r * stride + half * 8;
  union { v16bf v; v8bf h[2]; } u;
  u.h[0] = *reinterpret_cast<const v8bf*>(p);
  u.h[1] = *reinterpret_cast<const v8bf*>(p + 16);
  return u.v;
}

__device__ __forceinline__ v16bf load_b_frag(const bf16_t* __restrict__ base,
                                             int stride, int row, int lane) {
  int r = row + (lane & 15);
  int half = lane >> 4;
  const bf16_t* p = base + (size_t)r * stride + half * 8;
  union { v16bf v; v8bf h[2]; } u;
  u.h[0] = *reinterpret_cast<const v8bf*>(p);
  u.h[1] = *reinterpret_cast<const v8bf*>(p + 16);
  return u.v;
}

__device__ __forceinline__ v8f wmma_bf16(v16bf a, v16bf b, v8f c) {
  return __builtin_amdgcn_wmma_f32_16x16x32_bf16(false, a, false, b,
                                                 (short)0, c, false, false);
}

__device__ __forceinline__ float sigm(float x) {
  return 1.0f / (1.0f + __expf(-x));
}

// ---------------------------------------------------------------------------
// Preprocessing kernels: f32 -> bf16 convert / transpose-convert / copy
// ---------------------------------------------------------------------------
__global__ void cvt_bf16_kernel(const float* __restrict__ in,
                                bf16_t* __restrict__ out, long n) {
  long i = (long)blockIdx.x * 256 + threadIdx.x;
  if (i < n) out[i] = (bf16_t)in[i];
}

__global__ void transpose_bf16_kernel(const float* __restrict__ in,
                                      bf16_t* __restrict__ out, int K, int N) {
  long i = (long)blockIdx.x * 256 + threadIdx.x;
  if (i < (long)K * N) {
    int k = (int)(i / N);
    int n = (int)(i % N);
    out[(size_t)n * K + k] = (bf16_t)in[i];
  }
}

__global__ void copy_f32_kernel(const float* __restrict__ in,
                                float* __restrict__ out, long n) {
  long i = (long)blockIdx.x * 256 + threadIdx.x;
  if (i < n) out[i] = in[i];
}

// ---------------------------------------------------------------------------
// Generic bf16 WMMA GEMM: out[M,N](f32) = A[M,K](bf16) @ Bt[N,K]^T + bias[N]
// Block = 256 threads = 8 waves (2x4), each wave computes a 32x32 register
// tile (2 A-frags x 2 B-frags -> 4 WMMAs per K-chunk: 2x flops/byte vs 16x16).
// Block tile = 64(M) x 128(N).
// MODE 0: out row = m
// MODE 1: A rows are (b,t) [m = b*T + t]; out rows are (t,b)  (x@W -> Z)
// MODE 2: A rows are (t,b) [m = t*B + b]; out elem -> out[b][t*N + n] (fc1)
// ---------------------------------------------------------------------------
template <int MODE>
__device__ __forceinline__ void store_tile(float* __restrict__ out, v8f acc,
                                           int mbase, int nbase, int lane,
                                           int N, int P0, int P1, float bv) {
  int n = nbase + (lane & 15);
#pragma unroll
  for (int v = 0; v < 8; ++v) {
    int m = mbase + v + 8 * (lane >> 4);
    float val = acc[v] + bv;
    size_t idx;
    if (MODE == 0) {
      idx = (size_t)m * N + n;
    } else if (MODE == 1) {
      int bb = m / P1;  // P1 = T
      int tt = m % P1;
      idx = ((size_t)tt * P0 + bb) * N + n;  // Z[t][b][n]
    } else {
      int tt = m / P0;  // P0 = B
      int bb = m % P0;
      idx = (size_t)bb * ((size_t)P1 * N) + (size_t)tt * N + n;  // out[b][t*N+n]
    }
    out[idx] = val;
  }
}

template <int MODE>
__global__ __launch_bounds__(256) void gemm_bf16_kernel(
    const bf16_t* __restrict__ A, const bf16_t* __restrict__ Bt,
    const float* __restrict__ bias, float* __restrict__ out,
    int M, int N, int K, int P0, int P1) {
  int lane = threadIdx.x & 31;
  int wave = threadIdx.x >> 5;
  int wm = wave >> 2;   // 0..1
  int wn = wave & 3;    // 0..3
  int m0 = blockIdx.x * 64 + wm * 32;
  int n0 = blockIdx.y * 128 + wn * 32;

  v8f acc00 = {}, acc01 = {}, acc10 = {}, acc11 = {};
  for (int k0 = 0; k0 < K; k0 += 32) {
    v16bf a0 = load_a_frag(A + k0, K, m0, lane);
    v16bf a1 = load_a_frag(A + k0, K, m0 + 16, lane);
    v16bf b0 = load_b_frag(Bt + k0, K, n0, lane);
    v16bf b1 = load_b_frag(Bt + k0, K, n0 + 16, lane);
    acc00 = wmma_bf16(a0, b0, acc00);
    acc01 = wmma_bf16(a0, b1, acc01);
    acc10 = wmma_bf16(a1, b0, acc10);
    acc11 = wmma_bf16(a1, b1, acc11);
  }

  float bv0 = bias ? bias[n0 + (lane & 15)] : 0.0f;
  float bv1 = bias ? bias[n0 + 16 + (lane & 15)] : 0.0f;
  store_tile<MODE>(out, acc00, m0,      n0,      lane, N, P0, P1, bv0);
  store_tile<MODE>(out, acc01, m0,      n0 + 16, lane, N, P0, P1, bv1);
  store_tile<MODE>(out, acc10, m0 + 16, n0,      lane, N, P0, P1, bv0);
  store_tile<MODE>(out, acc11, m0 + 16, n0 + 16, lane, N, P0, P1, bv1);
}

// ---------------------------------------------------------------------------
// Fused LSTM step: gates = Zt + hprev @ U^T ; activations ; c,h update.
// This kernel is the sequential critical path (512 dependent launches), so it
// is built for latency, not just throughput: each block owns one 16(b)x16(n)
// tile; its 8 waves cover {4 gates} x {2 K-halves} (16-deep WMMA chains),
// partial sums are reduced through LDS, and the 256 threads then apply the
// activations + c/h update. Grid = (B/16, H/16) = 256 blocks = 2048 waves,
// spreading the step across the whole chip.
// ---------------------------------------------------------------------------
__global__ __launch_bounds__(256) void lstm_step_kernel(
    const bf16_t* __restrict__ hprev,  // (B,H) bf16
    const bf16_t* __restrict__ Ut,     // (4H,H) bf16 (transposed U)
    const float*  __restrict__ Zt,     // (B,4H) f32, bias already folded in
    float*        __restrict__ c,      // (B,H) f32, updated in place
    bf16_t*       __restrict__ hout)   // (B,H) bf16  (== ys[t])
{
  __shared__ float smem[8][32][8];
  int lane = threadIdx.x & 31;
  int wave = threadIdx.x >> 5;  // 0..7
  int g  = wave >> 1;           // gate 0..3 (i,f,g,o)
  int ks = wave & 1;            // K half
  int m0 = blockIdx.x * 16;
  int n0 = blockIdx.y * 16;

  v8f acc = {};
  const int kbeg = ks * (Hn / 2);
  const int kend = kbeg + (Hn / 2);
  for (int k0 = kbeg; k0 < kend; k0 += 32) {
    v16bf a = load_a_frag(hprev + k0, Hn, m0, lane);
    v16bf b = load_b_frag(Ut + k0, Hn, g * Hn + n0, lane);
    acc = wmma_bf16(a, b, acc);
  }
#pragma unroll
  for (int v = 0; v < 8; ++v) smem[wave][lane][v] = acc[v];
  __syncthreads();

  // Epilogue: thread tid owns element (m_local = tid/16, n_local = tid%16).
  int tid = threadIdx.x;
  int ml = tid >> 4;
  int nl = tid & 15;
  int l = nl + ((ml >> 3) << 4);  // lane holding this element in C-layout
  int v = ml & 7;                 // VGPR index within accumulator
  int m = m0 + ml;
  int n = n0 + nl;
  const float* z = Zt + (size_t)m * Gn + n;
  float gi = smem[0][l][v] + smem[1][l][v] + z[0 * Hn];
  float gf = smem[2][l][v] + smem[3][l][v] + z[1 * Hn];
  float gg = smem[4][l][v] + smem[5][l][v] + z[2 * Hn];
  float go = smem[6][l][v] + smem[7][l][v] + z[3 * Hn];
  size_t ci = (size_t)m * Hn + n;
  float cn = sigm(gf) * c[ci] + sigm(gi) * tanhf(gg);
  c[ci] = cn;
  hout[ci] = (bf16_t)(sigm(go) * tanhf(cn));
}

// ---------------------------------------------------------------------------
// fc2: out36[b][j] = sum_k fc1out[b][k] * w2[k][j] + b2[j], K = T*H/2 = 131072
// One block per batch row; 256 threads stride K, LDS-reduce 36 partials.
// ---------------------------------------------------------------------------
__global__ __launch_bounds__(256) void fc2_kernel(
    const float* __restrict__ fc1out, const float* __restrict__ w2,
    const float* __restrict__ b2, float* __restrict__ out36) {
  __shared__ float s[36];
  int b = blockIdx.x;
  int tid = threadIdx.x;
  if (tid < 36) s[tid] = 0.0f;
  __syncthreads();

  float acc[36];
#pragma unroll
  for (int j = 0; j < 36; ++j) acc[j] = 0.0f;

  const int Kf = Tn * (Hn / 2);
  const float* row = fc1out + (size_t)b * Kf;
  for (int k = tid; k < Kf; k += 256) {
    float a = row[k];
    const float* wr = w2 + (size_t)k * 36;
#pragma unroll
    for (int j = 0; j < 36; ++j) acc[j] += a * wr[j];
  }
#pragma unroll
  for (int j = 0; j < 36; ++j) atomicAdd(&s[j], acc[j]);
  __syncthreads();
  if (tid < 36) out36[b * 36 + tid] = s[tid] + b2[tid];
}

// fc3 + fc4 + relu: tiny, one block of 64 threads (one per batch row)
__global__ void fc34_kernel(const float* __restrict__ out36,
                            const float* __restrict__ w3, const float* __restrict__ b3,
                            const float* __restrict__ w4, const float* __restrict__ b4,
                            float* __restrict__ out) {
  int b = threadIdx.x;
  if (b >= Bn) return;
  float y3[6];
#pragma unroll
  for (int j = 0; j < 6; ++j) {
    float sacc = b3[j];
    for (int k = 0; k < 36; ++k) sacc += out36[b * 36 + k] * w3[k * 6 + j];
    y3[j] = sacc;
  }
#pragma unroll
  for (int j = 0; j < 6; ++j) {
    float sacc = b4[j];
#pragma unroll
    for (int k = 0; k < 6; ++k) sacc += y3[k] * w4[k * 6 + j];
    out[b * 6 + j] = sacc > 0.0f ? sacc : 0.0f;
  }
}

// ---------------------------------------------------------------------------
// Host-side orchestration
// ---------------------------------------------------------------------------
extern "C" void kernel_launch(void* const* d_in, const int* in_sizes, int n_in,
                              void* d_out, int out_size, void* d_ws, size_t ws_size,
                              hipStream_t stream) {
  const float* x     = (const float*)d_in[0];
  const float* h0    = (const float*)d_in[1];
  const float* c0    = (const float*)d_in[2];
  const float* W1    = (const float*)d_in[3];
  const float* U1    = (const float*)d_in[4];
  const float* b1    = (const float*)d_in[5];
  const float* W2    = (const float*)d_in[6];
  const float* U2    = (const float*)d_in[7];
  const float* b2    = (const float*)d_in[8];
  const float* fc1_w = (const float*)d_in[9];
  const float* fc1_b = (const float*)d_in[10];
  const float* fc2_w = (const float*)d_in[11];
  const float* fc2_b = (const float*)d_in[12];
  const float* fc3_w = (const float*)d_in[13];
  const float* fc3_b = (const float*)d_in[14];
  const float* fc4_w = (const float*)d_in[15];
  const float* fc4_b = (const float*)d_in[16];
  float* out = (float*)d_out;

  char* ws = (char*)d_ws;
  size_t off = 0;
  auto alloc = [&](size_t bytes) -> char* {
    char* p = ws + off;
    off = (off + bytes + 255) & ~(size_t)255;
    return p;
  };

  float*  Z       = (float*) alloc((size_t)Tn * Bn * Gn * 4);     // shared Z1/Z2
  bf16_t* xbf     = (bf16_t*)alloc((size_t)Bn * Tn * Dn * 2);
  bf16_t* w1t     = (bf16_t*)alloc((size_t)Gn * Dn * 2);
  bf16_t* u1t     = (bf16_t*)alloc((size_t)Gn * Hn * 2);
  bf16_t* w2t     = (bf16_t*)alloc((size_t)Gn * Hn * 2);
  bf16_t* u2t     = (bf16_t*)alloc((size_t)Gn * Hn * 2);
  bf16_t* fc1wt   = (bf16_t*)alloc((size_t)(Hn / 2) * Hn * 2);
  bf16_t* ys1bf   = (bf16_t*)alloc((size_t)Tn * Bn * Hn * 2);
  bf16_t* ys2bf   = (bf16_t*)alloc((size_t)Tn * Bn * Hn * 2);
  bf16_t* h0bf    = (bf16_t*)alloc((size_t)Bn * Hn * 2);
  float*  cbuf    = (float*) alloc((size_t)Bn * Hn * 4);
  float*  fc1out  = (float*) alloc((size_t)Bn * Tn * (Hn / 2) * 4);
  float*  out36   = (float*) alloc((size_t)Bn * 36 * 4);
  (void)ws_size; (void)in_sizes; (void)n_in; (void)out_size;

  const int M = Bn * Tn;  // 16384

  // ---- preprocessing: bf16 conversions / transposes -----------------------
  {
    long n = (long)Bn * Tn * Dn;
    cvt_bf16_kernel<<<(unsigned)((n + 255) / 256), 256, 0, stream>>>(x, xbf, n);
  }
  {
    long n = (long)Dn * Gn;
    transpose_bf16_kernel<<<(unsigned)((n + 255) / 256), 256, 0, stream>>>(W1, w1t, Dn, Gn);
  }
  {
    long n = (long)Hn * Gn;
    transpose_bf16_kernel<<<(unsigned)((n + 255) / 256), 256, 0, stream>>>(U1, u1t, Hn, Gn);
    transpose_bf16_kernel<<<(unsigned)((n + 255) / 256), 256, 0, stream>>>(W2, w2t, Hn, Gn);
    transpose_bf16_kernel<<<(unsigned)((n + 255) / 256), 256, 0, stream>>>(U2, u2t, Hn, Gn);
  }
  {
    long n = (long)Hn * (Hn / 2);
    transpose_bf16_kernel<<<(unsigned)((n + 255) / 256), 256, 0, stream>>>(fc1_w, fc1wt, Hn, Hn / 2);
  }
  {
    long n = (long)Bn * Hn;
    cvt_bf16_kernel<<<(unsigned)((n + 255) / 256), 256, 0, stream>>>(h0, h0bf, n);
    copy_f32_kernel<<<(unsigned)((n + 255) / 256), 256, 0, stream>>>(c0, cbuf, n);
  }

  // ---- Z1 = x @ W1 + b1, written as Z[t][b][4H] ---------------------------
  gemm_bf16_kernel<1><<<dim3(M / 64, Gn / 128), 256, 0, stream>>>(
      xbf, w1t, b1, Z, M, Gn, Dn, Bn, Tn);

  // ---- LSTM layer 1 (h double-buffers through ys1bf) ----------------------
  for (int t = 0; t < Tn; ++t) {
    const bf16_t* hprev = (t == 0) ? h0bf : (ys1bf + (size_t)(t - 1) * Bn * Hn);
    lstm_step_kernel<<<dim3(Bn / 16, Hn / 16), 256, 0, stream>>>(
        hprev, u1t, Z + (size_t)t * Bn * Gn, cbuf, ys1bf + (size_t)t * Bn * Hn);
  }

  // ---- Z2 = ys1 @ W2 + b2 (rows already (t,b) ordered) --------------------
  gemm_bf16_kernel<0><<<dim3(M / 64, Gn / 128), 256, 0, stream>>>(
      ys1bf, w2t, b2, Z, M, Gn, Hn, 0, 0);

  // ---- LSTM layer 2 (seeded with h1_final = ys1[T-1], c carried in place) -
  for (int t = 0; t < Tn; ++t) {
    const bf16_t* hprev = (t == 0) ? (ys1bf + (size_t)(Tn - 1) * Bn * Hn)
                                   : (ys2bf + (size_t)(t - 1) * Bn * Hn);
    lstm_step_kernel<<<dim3(Bn / 16, Hn / 16), 256, 0, stream>>>(
        hprev, u2t, Z + (size_t)t * Bn * Gn, cbuf, ys2bf + (size_t)t * Bn * Hn);
  }

  // ---- fc1: (B*T, H) @ (H, H/2), scatter to fc1out[b][t*(H/2)+n] ----------
  gemm_bf16_kernel<2><<<dim3(M / 64, (Hn / 2) / 128), 256, 0, stream>>>(
      ys2bf, fc1wt, fc1_b, fc1out, M, Hn / 2, Hn, Bn, Tn);

  // ---- fc2 / fc3 / fc4 + relu --------------------------------------------
  fc2_kernel<<<Bn, 256, 0, stream>>>(fc1out, fc2_w, fc2_b, out36);
  fc34_kernel<<<1, 64, 0, stream>>>(out36, fc3_w, fc3_b, fc4_w, fc4_b, out);
}
